// GraphAttentionLayer_20736102105164
// MI455X (gfx1250) — matrix operand order repaired and checked by reading, not compiled, not run on hardware
//
#include <hip/hip_runtime.h>
#include <hip/hip_bf16.h>
#include <stdint.h>

#define N_NODES 8192
#define F_IN    128
#define F_OUT   64
#define ALPHA   0.2f
#define LOG2E   1.4426950408889634f

typedef __attribute__((ext_vector_type(16))) __bf16       v16bf;
typedef __attribute__((ext_vector_type(8)))  float        v8f;
typedef __attribute__((ext_vector_type(8)))  unsigned int v8u;
typedef __attribute__((ext_vector_type(4)))  unsigned int u32x4;
typedef __attribute__((ext_vector_type(4)))  float        f32x4;

__device__ __forceinline__ unsigned short f2bf(float x) {
    // fp32 -> bf16 round-to-nearest-even (off hot path)
    unsigned int u = __float_as_uint(x);
    u += 0x7FFFu + ((u >> 16) & 1u);
    return (unsigned short)(u >> 16);
}

// ---------------------------------------------------------------------------
// Kernel 1: h = x @ W (fp32 [8192,64]) and hT = bf16(h)^T ([64][8192], the
// WMMA-B friendly column-major copy; total 1 MB -> stays resident in L2).
// ---------------------------------------------------------------------------
__global__ __launch_bounds__(256) void gat_h_kernel(const float* __restrict__ x,
                                                    const float* __restrict__ W,
                                                    float* __restrict__ h,
                                                    unsigned short* __restrict__ hT) {
    int idx = blockIdx.x * 256 + threadIdx.x;     // [0, 8192*64)
    int i = idx >> 6;
    int c = idx & 63;
    const float* xr = x + (size_t)i * F_IN;
    float acc = 0.f;
#pragma unroll 8
    for (int k = 0; k < F_IN; ++k) acc = fmaf(xr[k], W[k * F_OUT + c], acc);
    h[idx] = acc;
    hT[(size_t)c * N_NODES + i] = f2bf(acc);
}

// ---------------------------------------------------------------------------
// Kernel 2: f = log2(e) * (h @ a[:64]), g = log2(e) * (h @ a[64:]).
// Pre-scaling by log2(e) lets the main kernel use the raw v_exp_f32 (exp2)
// with no per-element multiply: leaky_relu commutes with a positive scale.
// ---------------------------------------------------------------------------
__global__ __launch_bounds__(256) void gat_fg_kernel(const float* __restrict__ h,
                                                     const float* __restrict__ a,
                                                     float* __restrict__ f,
                                                     float* __restrict__ g) {
    int i = blockIdx.x * 256 + threadIdx.x;
    if (i >= N_NODES) return;
    const float* hr = h + (size_t)i * F_OUT;
    float fa = 0.f, ga = 0.f;
#pragma unroll 8
    for (int c = 0; c < F_OUT; ++c) {
        float v = hr[c];
        fa = fmaf(v, a[c], fa);
        ga = fmaf(v, a[F_OUT + c], ga);
    }
    f[i] = fa * LOG2E;
    g[i] = ga * LOG2E;
}

// ---------------------------------------------------------------------------
// Main fused kernel: one pass over adj (256 MB = the 23.3 TB/s bandwidth
// floor, ~11 us). Block = 128 threads = 4 waves, one 16-row block per
// workgroup; wave w handles j-chunks jbase = w*32 + n*128.
// Per chunk: build P = adj * exp2(leaky_relu(f_i+g_j)) directly in the WMMA
// A-fragment layout (adj is exactly {0,1} so the mask is a multiply), then
// 4x v_wmma_f32_16x16x32_bf16 against resident hT fragments (B), fp32
// accumulators. Row denominators accumulated alongside (no softmax max
// needed: scores are O(1)). Cross-wave combine + divide via LDS.
// ---------------------------------------------------------------------------
__global__ __launch_bounds__(128) void gat_attn_kernel(const float* __restrict__ adj,
                                                       const float* __restrict__ f,
                                                       const float* __restrict__ g,
                                                       const unsigned short* __restrict__ hT,
                                                       float* __restrict__ out) {
    __shared__ float lds_num[4][16 * F_OUT];   // 16 KB
    __shared__ float lds_den[4][32];

    const int rbase = blockIdx.x * 16;
    const int lane  = threadIdx.x & 31;
    const int wid   = threadIdx.x >> 5;
    const int row   = lane & 15;               // A row / B column / D column
    const int hi    = lane >> 4;               // K-half select

    const float  frow    = f[rbase + row];     // already scaled by log2(e)
    const float* adj_row = adj + (size_t)(rbase + row) * N_NODES;

    v8f zero = {};
    v8f acc[4];
#pragma unroll
    for (int t = 0; t < 4; ++t) acc[t] = zero;
    float denom = 0.f;

    for (int jbase = wid * 32; jbase < N_NODES; jbase += 128) {
        // A-fragment K layout (16-bit A 16x32): lanes hi=0 -> K {0..7,16..23},
        // lanes hi=1 -> K {8..15,24..31}; pairs packed low-first.
        const int j0 = jbase + hi * 8;         // K group for VGPRs 0..3
        const int j1 = jbase + 16 + hi * 8;    // K group for VGPRs 4..7

        // adj is streamed exactly once: non-temporal so hT stays hot in L2.
        f32x4 A0 = __builtin_nontemporal_load((const f32x4*)(adj_row + j0));
        f32x4 A1 = __builtin_nontemporal_load((const f32x4*)(adj_row + j0 + 4));
        f32x4 A2 = __builtin_nontemporal_load((const f32x4*)(adj_row + j1));
        f32x4 A3 = __builtin_nontemporal_load((const f32x4*)(adj_row + j1 + 4));
        f32x4 G0 = *(const f32x4*)(g + j0);
        f32x4 G1 = *(const f32x4*)(g + j0 + 4);
        f32x4 G2 = *(const f32x4*)(g + j1);
        f32x4 G3 = *(const f32x4*)(g + j1 + 4);

        if (jbase + 160 <= N_NODES)
            __builtin_prefetch(adj_row + j0 + 128, 0, 1);

        float ax[16] = {A0[0],A0[1],A0[2],A0[3], A1[0],A1[1],A1[2],A1[3],
                        A2[0],A2[1],A2[2],A2[3], A3[0],A3[1],A3[2],A3[3]};
        float gx[16] = {G0[0],G0[1],G0[2],G0[3], G1[0],G1[1],G1[2],G1[3],
                        G2[0],G2[1],G2[2],G2[3], G3[0],G3[1],G3[2],G3[3]};

        float p[16];
#pragma unroll
        for (int q = 0; q < 16; ++q) {
            float t  = frow + gx[q];                    // log2-domain score
            float s  = fmaxf(t, ALPHA * t);             // leaky_relu (alpha<1)
            float pv = ax[q] * __builtin_amdgcn_exp2f(s); // adj in {0,1}: mask==mul
            denom += pv;
            p[q] = pv;
        }

        v16bf ab;
#pragma unroll
        for (int q = 0; q < 16; ++q) ab[q] = (__bf16)p[q];

        // B fragments: 32x16 bf16, lane owns col row=lane%16, K pairs in
        // consecutive dwords; hT is column-major so this is 2 contiguous b128.
        const int jb = jbase + hi * 8;
#pragma unroll
        for (int t = 0; t < 4; ++t) {
            const unsigned short* hp = hT + (size_t)(t * 16 + row) * N_NODES + jb;
            union { v8u u; v16bf b; u32x4 q[2]; } bfrag;
            bfrag.q[0] = *(const u32x4*)(hp);
            bfrag.q[1] = *(const u32x4*)(hp + 16);
            acc[t] = __builtin_amdgcn_wmma_f32_16x16x32_bf16(
                false, ab, false, bfrag.b, (short)0, acc[t], false, false);
        }
    }

    // D layout: (VGPR v, lane) -> row v + 8*hi, col t*16 + row
#pragma unroll
    for (int t = 0; t < 4; ++t)
#pragma unroll
        for (int v = 0; v < 8; ++v)
            lds_num[wid][(v + 8 * hi) * F_OUT + t * 16 + row] = acc[t][v];
    lds_den[wid][lane] = denom;
    __syncthreads();

    for (int e = threadIdx.x; e < 16 * F_OUT; e += 128) {
        int r = e >> 6;
        float num = lds_num[0][e] + lds_num[1][e] + lds_num[2][e] + lds_num[3][e];
        float den = 0.f;
#pragma unroll
        for (int w = 0; w < 4; ++w) den += lds_den[w][r] + lds_den[w][r + 16];
        out[(size_t)rbase * F_OUT + e] = num / den;
    }
}

// ---------------------------------------------------------------------------
extern "C" void kernel_launch(void* const* d_in, const int* in_sizes, int n_in,
                              void* d_out, int out_size, void* d_ws, size_t ws_size,
                              hipStream_t stream) {
    const float* adj = (const float*)d_in[0];
    const float* x   = (const float*)d_in[1];
    const float* W   = (const float*)d_in[2];
    const float* a   = (const float*)d_in[3];
    float* out = (float*)d_out;

    char* ws = (char*)d_ws;
    float*          h  = (float*)ws;                                          // 2 MB
    unsigned short* hT = (unsigned short*)(ws + (size_t)N_NODES * F_OUT * 4); // 1 MB
    float*          f  = (float*)(ws + 3u * 1024u * 1024u);                   // 32 KB
    float*          gv = (float*)(ws + 3u * 1024u * 1024u + N_NODES * 4);     // 32 KB

    gat_h_kernel<<<(N_NODES * F_OUT) / 256, 256, 0, stream>>>(x, W, h, hT);
    gat_fg_kernel<<<N_NODES / 256, 256, 0, stream>>>(h, a, f, gv);
    gat_attn_kernel<<<N_NODES / 16, 128, 0, stream>>>(adj, f, gv, hT, out);
}
